// EncoderLayer_50130858279191
// MI455X (gfx1250) — matrix operand order repaired and checked by reading, not compiled, not run on hardware
//
#include <hip/hip_runtime.h>
#include <hip/hip_bf16.h>
#include <math.h>

// Problem constants (from the reference)
#define NN 50000
#define EE 600000
#define DD 128
#define DFF 512

typedef __bf16 bf16;
typedef __bf16 bf16x16 __attribute__((ext_vector_type(16)));
typedef __bf16 bf16x8  __attribute__((ext_vector_type(8)));
typedef float  f32x8   __attribute__((ext_vector_type(8)));

// ---------------------------------------------------------------------------
// Cooperative stage: MT x 128 fp32 tile -> LDS bf16 (16B stores)
// 256 threads: per pass, thread t -> row (t>>4)+16*pass, cols (t&15)*8 .. +7
// ---------------------------------------------------------------------------
template <int MT>
static __device__ inline void stage_a_tile(const float* __restrict__ A, int m0,
                                           bf16* __restrict__ aLds)
{
#pragma unroll
    for (int pass = 0; pass < MT / 16; ++pass) {
        const int row = (threadIdx.x >> 4) + pass * 16;
        const int c8  = (threadIdx.x & 15) * 8;
        const float* p = A + (size_t)(m0 + row) * DD + c8;
        const float4 u = *(const float4*)p;
        const float4 w = *(const float4*)(p + 4);
        bf16x8 pk;
        pk[0] = (bf16)u.x; pk[1] = (bf16)u.y; pk[2] = (bf16)u.z; pk[3] = (bf16)u.w;
        pk[4] = (bf16)w.x; pk[5] = (bf16)w.y; pk[6] = (bf16)w.z; pk[7] = (bf16)w.w;
        *(bf16x8*)(aLds + row * DD + c8) = pk;
    }
}

// A-fragment from an LDS-resident bf16 row (16-bit A 16x32, contiguous k order used by this code):
// lane r(0..15)=row, half=lane>>4:
//   elems e (0..15) -> K = k0+half*16+e
static __device__ inline bf16x16 a_frag(const bf16* __restrict__ Ar, int k0, int half)
{
    bf16x16 a;
    const bf16x8 lo = *(const bf16x8*)(Ar + k0 + half * 16);
    const bf16x8 hi = *(const bf16x8*)(Ar + k0 + half * 16 + 8);
#pragma unroll
    for (int e = 0; e < 8; ++e) { a[e] = lo[e]; a[e + 8] = hi[e]; }
    return a;
}

// One 16x16 WMMA tile, A from LDS (bf16, row stride lda), B from bf16 Wt[Dout][Din]
// B-fragment (16-bit B 32x16): lane r = column, elems e -> K = k0+half*16+e
static __device__ inline f32x8 tile_gemm_lds(const bf16* __restrict__ aLds, int lda,
                                             const bf16* __restrict__ Wt, int ldw,
                                             int n0, int Din)
{
    const int lane = threadIdx.x & 31;
    const int r    = lane & 15;
    const int half = lane >> 4;
    const bf16* Ar = aLds + (size_t)r * lda;
    const bf16* Bc = Wt + (size_t)(n0 + r) * ldw;
    f32x8 acc = {0.f, 0.f, 0.f, 0.f, 0.f, 0.f, 0.f, 0.f};
    for (int k0 = 0; k0 < Din; k0 += 32) {
        const bf16x16 a = a_frag(Ar, k0, half);
        const bf16x16 b = *(const bf16x16*)(Bc + k0 + half * 16);
        acc = __builtin_amdgcn_wmma_f32_16x16x32_bf16(false, a, false, b,
                                                      (short)0, acc, false, false);
    }
    return acc;
}

// 4 row-tiles (64 rows) x one 16-col tile: B fragment loaded once per k-step,
// reused across 4 WMMAs -> 4x less weight traffic.
static __device__ inline void tile_gemm_lds_m64(const bf16* __restrict__ aLds, int lda,
                                                const bf16* __restrict__ Wt, int ldw,
                                                int n0, int Din, f32x8 acc[4])
{
    const int lane = threadIdx.x & 31;
    const int r    = lane & 15;
    const int half = lane >> 4;
    const bf16* Bc = Wt + (size_t)(n0 + r) * ldw;
    for (int k0 = 0; k0 < Din; k0 += 32) {
        const bf16x16 b = *(const bf16x16*)(Bc + k0 + half * 16);
#pragma unroll
        for (int rt = 0; rt < 4; ++rt) {
            const bf16x16 a = a_frag(aLds + (size_t)(rt * 16 + r) * lda, k0, half);
            acc[rt] = __builtin_amdgcn_wmma_f32_16x16x32_bf16(false, a, false, b,
                                                              (short)0, acc[rt],
                                                              false, false);
        }
    }
}

// Parallel LayerNorm over a 16x128 LDS tile: 16 threads/row, 8 cols each.
static __device__ inline void ln_tail16(float (*t)[DD + 1], int m0,
                                        const float* __restrict__ g,
                                        const float* __restrict__ beta,
                                        float* __restrict__ out)
{
    const int row = threadIdx.x >> 4;
    const int seg = (threadIdx.x & 15) * 8;
    float s = 0.f;
#pragma unroll
    for (int i = 0; i < 8; ++i) s += t[row][seg + i];
#pragma unroll
    for (int o = 8; o >= 1; o >>= 1) s += __shfl_xor(s, o, 16);
    const float mu = s * (1.f / DD);
    float v = 0.f;
#pragma unroll
    for (int i = 0; i < 8; ++i) { float d = t[row][seg + i] - mu; v += d * d; }
#pragma unroll
    for (int o = 8; o >= 1; o >>= 1) v += __shfl_xor(v, o, 16);
    const float rs = rsqrtf(v * (1.f / DD) + 1e-5f);
    float* op = out + (size_t)(m0 + row) * DD + seg;
#pragma unroll
    for (int i = 0; i < 8; ++i)
        op[i] = (t[row][seg + i] - mu) * rs * g[seg + i] + beta[seg + i];
}

// Parallel LayerNorm over a 64x128 LDS tile: 4 threads/row, 32 cols each.
static __device__ inline void ln_tail64(float (*t)[DD + 1], int m0,
                                        const float* __restrict__ g,
                                        const float* __restrict__ beta,
                                        float* __restrict__ out)
{
    const int row = threadIdx.x >> 2;
    const int seg = (threadIdx.x & 3) * 32;
    float s = 0.f;
#pragma unroll
    for (int i = 0; i < 32; ++i) s += t[row][seg + i];
#pragma unroll
    for (int o = 2; o >= 1; o >>= 1) s += __shfl_xor(s, o, 4);
    const float mu = s * (1.f / DD);
    float v = 0.f;
#pragma unroll
    for (int i = 0; i < 32; ++i) { float d = t[row][seg + i] - mu; v += d * d; }
#pragma unroll
    for (int o = 2; o >= 1; o >>= 1) v += __shfl_xor(v, o, 4);
    const float rs = rsqrtf(v * (1.f / DD) + 1e-5f);
    float* op = out + (size_t)(m0 + row) * DD + seg;
#pragma unroll
    for (int i = 0; i < 32; ++i)
        op[i] = (t[row][seg + i] - mu) * rs * g[seg + i] + beta[seg + i];
}

// --------------------------- weight transpose + cvt ------------------------
__global__ void k_tcvt(const float* __restrict__ W, bf16* __restrict__ Wt,
                       int Din, int Dout)
{
    int idx = blockIdx.x * blockDim.x + threadIdx.x;
    int i = idx / Dout;
    int o = idx % Dout;
    Wt[(size_t)o * Din + i] = (bf16)W[(size_t)i * Dout + o];
}

__global__ void k_fill0(float* __restrict__ p)
{
    size_t i = (size_t)blockIdx.x * blockDim.x + threadIdx.x;
    p[i] = 0.f;
}

// ------------- fused q/k/v projection: x read once, 3 WMMAs per k-step ------
__global__ void k_gemm_qkv(const float* __restrict__ X,
                           const bf16* __restrict__ WqT, const float* __restrict__ bq,
                           const bf16* __restrict__ WkT, const float* __restrict__ bk,
                           const bf16* __restrict__ WvT, const float* __restrict__ bv,
                           float* __restrict__ Q, float* __restrict__ K,
                           float* __restrict__ V)
{
    __shared__ __align__(16) bf16 aLds[16 * DD];
    const int m0 = blockIdx.x * 16;
    stage_a_tile<16>(X, m0, aLds);
    __syncthreads();

    const int wave = threadIdx.x >> 5;
    const int lane = threadIdx.x & 31;
    const int r    = lane & 15;
    const int half = lane >> 4;
    const int n0   = wave * 16;

    const bf16* Ar  = aLds + (size_t)r * DD;
    const bf16* Bq  = WqT + (size_t)(n0 + r) * DD;
    const bf16* Bk  = WkT + (size_t)(n0 + r) * DD;
    const bf16* Bv  = WvT + (size_t)(n0 + r) * DD;
    f32x8 cq = {0.f, 0.f, 0.f, 0.f, 0.f, 0.f, 0.f, 0.f};
    f32x8 ck = cq, cv = cq;
    for (int k0 = 0; k0 < DD; k0 += 32) {
        const bf16x16 a  = a_frag(Ar, k0, half);
        const bf16x16 b0 = *(const bf16x16*)(Bq + k0 + half * 16);
        const bf16x16 b1 = *(const bf16x16*)(Bk + k0 + half * 16);
        const bf16x16 b2 = *(const bf16x16*)(Bv + k0 + half * 16);
        cq = __builtin_amdgcn_wmma_f32_16x16x32_bf16(false, a, false, b0, (short)0, cq, false, false);
        ck = __builtin_amdgcn_wmma_f32_16x16x32_bf16(false, a, false, b1, (short)0, ck, false, false);
        cv = __builtin_amdgcn_wmma_f32_16x16x32_bf16(false, a, false, b2, (short)0, cv, false, false);
    }
    const float bbq = bq[n0 + r], bbk = bk[n0 + r], bbv = bv[n0 + r];
#pragma unroll
    for (int j = 0; j < 8; ++j) {
        const size_t o = (size_t)(m0 + j + half * 8) * DD + n0 + r;
        Q[o] = cq[j] + bbq;
        K[o] = ck[j] + bbk;
        V[o] = cv[j] + bbv;
    }
}

// --------- M=64 GEMM (Din = Dout = 128) + bias  (edge rows: E % 64 == 0) ----
__global__ void k_gemm128_m64(const float* __restrict__ A, const bf16* __restrict__ Wt,
                              const float* __restrict__ bias, float* __restrict__ out)
{
    __shared__ __align__(16) bf16 aLds[64 * DD];
    const int m0 = blockIdx.x * 64;
    stage_a_tile<64>(A, m0, aLds);
    __syncthreads();

    const int wave = threadIdx.x >> 5;
    const int lane = threadIdx.x & 31;
    const int r    = lane & 15;
    const int half = lane >> 4;
    const int n0   = wave * 16;
    f32x8 acc[4] = {{0.f,0.f,0.f,0.f,0.f,0.f,0.f,0.f},
                    {0.f,0.f,0.f,0.f,0.f,0.f,0.f,0.f},
                    {0.f,0.f,0.f,0.f,0.f,0.f,0.f,0.f},
                    {0.f,0.f,0.f,0.f,0.f,0.f,0.f,0.f}};
    tile_gemm_lds_m64(aLds, DD, Wt, DD, n0, DD, acc);
    const float bb = bias[n0 + r];
#pragma unroll
    for (int rt = 0; rt < 4; ++rt)
#pragma unroll
        for (int j = 0; j < 8; ++j)
            out[(size_t)(m0 + rt * 16 + j + half * 8) * DD + n0 + r] = acc[rt][j] + bb;
}

// -------- M=16 GEMM + bias + residual + LayerNorm (node side, N % 16) -------
__global__ void k_gemm_res_ln(const float* __restrict__ A, const bf16* __restrict__ Wt,
                              const float* __restrict__ bias, const float* __restrict__ res,
                              const float* __restrict__ g, const float* __restrict__ beta,
                              float* __restrict__ out)
{
    __shared__ __align__(16) bf16 aLds[16 * DD];
    __shared__ float t[16][DD + 1];
    const int m0 = blockIdx.x * 16;
    stage_a_tile<16>(A, m0, aLds);
    __syncthreads();

    const int wave = threadIdx.x >> 5;
    const int lane = threadIdx.x & 31;
    const int r    = lane & 15;
    const int half = lane >> 4;
    const int n0   = wave * 16;
    f32x8 acc = tile_gemm_lds(aLds, DD, Wt, DD, n0, DD);
    const float bb = bias[n0 + r];
#pragma unroll
    for (int j = 0; j < 8; ++j) {
        const int ml = j + half * 8;
        t[ml][n0 + r] = acc[j] + bb + res[(size_t)(m0 + ml) * DD + n0 + r];
    }
    __syncthreads();
    ln_tail16(t, m0, g, beta, out);
}

// -------- M=64 GEMM + bias + residual + LayerNorm (edge side) ---------------
__global__ void k_gemm_res_ln_m64(const float* __restrict__ A, const bf16* __restrict__ Wt,
                                  const float* __restrict__ bias, const float* __restrict__ res,
                                  const float* __restrict__ g, const float* __restrict__ beta,
                                  float* __restrict__ out)
{
    __shared__ __align__(16) bf16 aLds[64 * DD];
    __shared__ float t[64][DD + 1];
    const int m0 = blockIdx.x * 64;
    stage_a_tile<64>(A, m0, aLds);
    __syncthreads();

    const int wave = threadIdx.x >> 5;
    const int lane = threadIdx.x & 31;
    const int r    = lane & 15;
    const int half = lane >> 4;
    const int n0   = wave * 16;
    f32x8 acc[4] = {{0.f,0.f,0.f,0.f,0.f,0.f,0.f,0.f},
                    {0.f,0.f,0.f,0.f,0.f,0.f,0.f,0.f},
                    {0.f,0.f,0.f,0.f,0.f,0.f,0.f,0.f},
                    {0.f,0.f,0.f,0.f,0.f,0.f,0.f,0.f}};
    tile_gemm_lds_m64(aLds, DD, Wt, DD, n0, DD, acc);
    const float bb = bias[n0 + r];
#pragma unroll
    for (int rt = 0; rt < 4; ++rt)
#pragma unroll
        for (int j = 0; j < 8; ++j) {
            const int ml = rt * 16 + j + half * 8;
            t[ml][n0 + r] = acc[rt][j] + bb + res[(size_t)(m0 + ml) * DD + n0 + r];
        }
    __syncthreads();
    ln_tail64(t, m0, g, beta, out);
}

// -- M=16 fused FFN: y = LN(x + relu(x@W1+b1)@W2+b2), hidden in LDS (nodes) --
__global__ void k_ffn_ln(const float* __restrict__ Xin,
                         const bf16* __restrict__ W1t, const float* __restrict__ b1,
                         const bf16* __restrict__ W2t, const float* __restrict__ b2,
                         const float* __restrict__ g, const float* __restrict__ beta,
                         float* __restrict__ out)
{
    __shared__ __align__(16) bf16 aLds[16 * DD];
    __shared__ __align__(16) bf16 h[16 * DFF];
    __shared__ float t[16][DD + 1];
    const int m0 = blockIdx.x * 16;
    stage_a_tile<16>(Xin, m0, aLds);
    __syncthreads();

    const int wave = threadIdx.x >> 5;
    const int lane = threadIdx.x & 31;
    const int r    = lane & 15;
    const int half = lane >> 4;

#pragma unroll
    for (int i = 0; i < 4; ++i) {
        const int n0 = (wave * 4 + i) * 16;
        f32x8 acc = tile_gemm_lds(aLds, DD, W1t, DD, n0, DD);
        const float bb = b1[n0 + r];
#pragma unroll
        for (int j = 0; j < 8; ++j) {
            float v = acc[j] + bb;
            v = v > 0.f ? v : 0.f;
            h[(j + half * 8) * DFF + n0 + r] = (bf16)v;
        }
    }
    __syncthreads();

    const int n0 = wave * 16;
    f32x8 acc = tile_gemm_lds(h, DFF, W2t, DFF, n0, DFF);
    const float bb = b2[n0 + r];
#pragma unroll
    for (int j = 0; j < 8; ++j) {
        const int ml = j + half * 8;
        t[ml][n0 + r] = acc[j] + bb + Xin[(size_t)(m0 + ml) * DD + n0 + r];
    }
    __syncthreads();
    ln_tail16(t, m0, g, beta, out);
}

// -- M=64 fused FFN (edges): 4x B-fragment reuse in both GEMM stages ---------
__global__ void k_ffn_ln_m64(const float* __restrict__ Xin,
                             const bf16* __restrict__ W1t, const float* __restrict__ b1,
                             const bf16* __restrict__ W2t, const float* __restrict__ b2,
                             const float* __restrict__ g, const float* __restrict__ beta,
                             float* __restrict__ out)
{
    __shared__ __align__(16) bf16 aLds[64 * DD];    // 16 KB
    __shared__ __align__(16) bf16 h[64 * DFF];      // 64 KB
    __shared__ float t[64][DD + 1];                 // ~33 KB
    const int m0 = blockIdx.x * 64;
    stage_a_tile<64>(Xin, m0, aLds);
    __syncthreads();

    const int wave = threadIdx.x >> 5;
    const int lane = threadIdx.x & 31;
    const int r    = lane & 15;
    const int half = lane >> 4;

    // Stage 1: hidden = relu(X @ W1 + b1)  [64 x 512]
#pragma unroll
    for (int i = 0; i < 4; ++i) {
        const int n0 = (wave * 4 + i) * 16;
        f32x8 acc[4] = {{0.f,0.f,0.f,0.f,0.f,0.f,0.f,0.f},
                        {0.f,0.f,0.f,0.f,0.f,0.f,0.f,0.f},
                        {0.f,0.f,0.f,0.f,0.f,0.f,0.f,0.f},
                        {0.f,0.f,0.f,0.f,0.f,0.f,0.f,0.f}};
        tile_gemm_lds_m64(aLds, DD, W1t, DD, n0, DD, acc);
        const float bb = b1[n0 + r];
#pragma unroll
        for (int rt = 0; rt < 4; ++rt)
#pragma unroll
            for (int j = 0; j < 8; ++j) {
                float v = acc[rt][j] + bb;
                v = v > 0.f ? v : 0.f;
                h[(rt * 16 + j + half * 8) * DFF + n0 + r] = (bf16)v;
            }
    }
    __syncthreads();

    // Stage 2: y = hidden @ W2 + b2 + X (residual), Din = 512 from LDS
    const int n0 = wave * 16;
    f32x8 acc[4] = {{0.f,0.f,0.f,0.f,0.f,0.f,0.f,0.f},
                    {0.f,0.f,0.f,0.f,0.f,0.f,0.f,0.f},
                    {0.f,0.f,0.f,0.f,0.f,0.f,0.f,0.f},
                    {0.f,0.f,0.f,0.f,0.f,0.f,0.f,0.f}};
    tile_gemm_lds_m64(h, DFF, W2t, DFF, n0, DFF, acc);
    const float bb = b2[n0 + r];
#pragma unroll
    for (int rt = 0; rt < 4; ++rt)
#pragma unroll
        for (int j = 0; j < 8; ++j) {
            const int ml = rt * 16 + j + half * 8;
            t[ml][n0 + r] = acc[rt][j] + bb + Xin[(size_t)(m0 + ml) * DD + n0 + r];
        }
    __syncthreads();
    ln_tail64(t, m0, g, beta, out);
}

// --------- edge pass 1: m = exp(k[src]*q[dst]/4); me = m*eh; z[dst]+=m -------
__global__ void k_edge1(const float* __restrict__ q, const float* __restrict__ k,
                        float* __restrict__ me,          // in: eh, out: me (in place)
                        const int* __restrict__ src, const int* __restrict__ dst,
                        float* __restrict__ z)
{
    const size_t gid  = (size_t)blockIdx.x * blockDim.x + threadIdx.x;
    const size_t eidx = gid >> 5;
    const int    c    = (int)(gid & 31) * 4;
    const int s = src[eidx], d = dst[eidx];
    const float4 kk = *(const float4*)(k + (size_t)s * DD + c);
    const float4 qq = *(const float4*)(q + (size_t)d * DD + c);
    float4 m;
    m.x = __expf(kk.x * qq.x * 0.25f);
    m.y = __expf(kk.y * qq.y * 0.25f);
    m.z = __expf(kk.z * qq.z * 0.25f);
    m.w = __expf(kk.w * qq.w * 0.25f);
    float4* mep = (float4*)(me + eidx * DD + c);
    const float4 eh = *mep;
    *mep = make_float4(m.x * eh.x, m.y * eh.y, m.z * eh.z, m.w * eh.w);
    float* zp = z + (size_t)d * DD + c;
    atomicAdd(zp + 0, m.x);
    atomicAdd(zp + 1, m.y);
    atomicAdd(zp + 2, m.z);
    atomicAdd(zp + 3, m.w);
}

// --------- edge pass 2: xagg[dst] += me * v[src] / z[dst] -------------------
__global__ void k_edge2(const float* __restrict__ me, const float* __restrict__ v,
                        const float* __restrict__ z,
                        const int* __restrict__ src, const int* __restrict__ dst,
                        float* __restrict__ xagg)
{
    const size_t gid  = (size_t)blockIdx.x * blockDim.x + threadIdx.x;
    const size_t eidx = gid >> 5;
    const int    c    = (int)(gid & 31) * 4;
    const int s = src[eidx], d = dst[eidx];
    const float4 mm = *(const float4*)(me + eidx * DD + c);
    const float4 vv = *(const float4*)(v + (size_t)s * DD + c);
    const float4 zz = *(const float4*)(z + (size_t)d * DD + c);
    float* xp = xagg + (size_t)d * DD + c;
    atomicAdd(xp + 0, mm.x * vv.x / zz.x);
    atomicAdd(xp + 1, mm.y * vv.y / zz.y);
    atomicAdd(xp + 2, mm.z * vv.z / zz.z);
    atomicAdd(xp + 3, mm.w * vv.w / zz.w);
}

// ---------------------------------------------------------------------------
extern "C" void kernel_launch(void* const* d_in, const int* in_sizes, int n_in,
                              void* d_out, int out_size, void* d_ws, size_t ws_size,
                              hipStream_t stream)
{
    const float* x   = (const float*)d_in[0];
    const float* e   = (const float*)d_in[1];
    const int*   src = (const int*)d_in[2];
    const int*   dst = (const int*)d_in[3];
    const float* Wq  = (const float*)d_in[4];  const float* bq  = (const float*)d_in[5];
    const float* Wk  = (const float*)d_in[6];  const float* bk  = (const float*)d_in[7];
    const float* Wv  = (const float*)d_in[8];  const float* bv  = (const float*)d_in[9];
    const float* We  = (const float*)d_in[10]; const float* be  = (const float*)d_in[11];
    const float* Wox = (const float*)d_in[12]; const float* box = (const float*)d_in[13];
    const float* Woe = (const float*)d_in[14]; const float* boe = (const float*)d_in[15];
    const float* Wf1x = (const float*)d_in[16]; const float* bf1x = (const float*)d_in[17];
    const float* Wf2x = (const float*)d_in[18]; const float* bf2x = (const float*)d_in[19];
    const float* Wf1e = (const float*)d_in[20]; const float* bf1e = (const float*)d_in[21];
    const float* Wf2e = (const float*)d_in[22]; const float* bf2e = (const float*)d_in[23];
    const float* gx0 = (const float*)d_in[24]; const float* bx0 = (const float*)d_in[25];
    const float* gx1 = (const float*)d_in[26]; const float* bx1 = (const float*)d_in[27];
    const float* ge0 = (const float*)d_in[28]; const float* be0 = (const float*)d_in[29];
    const float* ge1 = (const float*)d_in[30]; const float* be1 = (const float*)d_in[31];

    float* out_x = (float*)d_out;                       // [N, D]
    float* out_e = out_x + (size_t)NN * DD;             // [E, D]

    // ---- workspace carve-up (256B aligned) ----
    char*  ws  = (char*)d_ws;
    size_t off = 0;
    auto carve = [&](size_t bytes) -> char* {
        char* p = ws + off;
        off += (bytes + 255) & ~(size_t)255;
        return p;
    };
    bf16* wqT  = (bf16*)carve((size_t)DD * DD * 2);
    bf16* wkT  = (bf16*)carve((size_t)DD * DD * 2);
    bf16* wvT  = (bf16*)carve((size_t)DD * DD * 2);
    bf16* weT  = (bf16*)carve((size_t)DD * DD * 2);
    bf16* woxT = (bf16*)carve((size_t)DD * DD * 2);
    bf16* woeT = (bf16*)carve((size_t)DD * DD * 2);
    bf16* wf1xT = (bf16*)carve((size_t)DD * DFF * 2);
    bf16* wf2xT = (bf16*)carve((size_t)DFF * DD * 2);
    bf16* wf1eT = (bf16*)carve((size_t)DD * DFF * 2);
    bf16* wf2eT = (bf16*)carve((size_t)DFF * DD * 2);
    float* qb   = (float*)carve((size_t)NN * DD * 4);
    float* kb   = (float*)carve((size_t)NN * DD * 4);
    float* vb   = (float*)carve((size_t)NN * DD * 4);
    float* zb   = (float*)carve((size_t)NN * DD * 4);
    float* xagg = (float*)carve((size_t)NN * DD * 4);
    float* x1   = (float*)carve((size_t)NN * DD * 4);
    float* meb  = (float*)carve((size_t)EE * DD * 4);   // eh -> me (in place)
    float* e1   = (float*)carve((size_t)EE * DD * 4);
    (void)ws_size; (void)n_in; (void)in_sizes; (void)out_size;

    // ---- 1) weights -> bf16 transposed ----
    k_tcvt<<<(DD * DD) / 256, 256, 0, stream>>>(Wq,  wqT,  DD, DD);
    k_tcvt<<<(DD * DD) / 256, 256, 0, stream>>>(Wk,  wkT,  DD, DD);
    k_tcvt<<<(DD * DD) / 256, 256, 0, stream>>>(Wv,  wvT,  DD, DD);
    k_tcvt<<<(DD * DD) / 256, 256, 0, stream>>>(We,  weT,  DD, DD);
    k_tcvt<<<(DD * DD) / 256, 256, 0, stream>>>(Wox, woxT, DD, DD);
    k_tcvt<<<(DD * DD) / 256, 256, 0, stream>>>(Woe, woeT, DD, DD);
    k_tcvt<<<(DD * DFF) / 256, 256, 0, stream>>>(Wf1x, wf1xT, DD, DFF);
    k_tcvt<<<(DFF * DD) / 256, 256, 0, stream>>>(Wf2x, wf2xT, DFF, DD);
    k_tcvt<<<(DD * DFF) / 256, 256, 0, stream>>>(Wf1e, wf1eT, DD, DFF);
    k_tcvt<<<(DFF * DD) / 256, 256, 0, stream>>>(Wf2e, wf2eT, DFF, DD);

    // ---- 2) projections (WMMA): q,k,v fused (x read once), eh (M=64) ----
    k_gemm_qkv<<<NN / 16, 256, 0, stream>>>(x, wqT, bq, wkT, bk, wvT, bv, qb, kb, vb);
    k_gemm128_m64<<<EE / 64, 256, 0, stream>>>(e, weT, be, meb);   // eh

    // ---- 3) zero z / xagg ----
    k_fill0<<<(NN * DD) / 256, 256, 0, stream>>>(zb);
    k_fill0<<<(NN * DD) / 256, 256, 0, stream>>>(xagg);

    // ---- 4) edge passes ----
    const int eblocks = (EE * (DD / 4)) / 256;   // one thread per float4
    k_edge1<<<eblocks, 256, 0, stream>>>(qb, kb, meb, src, dst, zb);
    k_edge2<<<eblocks, 256, 0, stream>>>(meb, vb, zb, src, dst, xagg);

    // ---- 5) attention output projections + residual + LN (WMMA, fused) ----
    k_gemm_res_ln<<<NN / 16, 256, 0, stream>>>(xagg, woxT, box, x, gx0, bx0, x1);
    k_gemm_res_ln_m64<<<EE / 64, 256, 0, stream>>>(meb, woeT, boe, e, ge0, be0, e1);

    // ---- 6) fused FFN + residual + LN -> final outputs (WMMA) ----
    k_ffn_ln<<<NN / 16, 256, 0, stream>>>(x1, wf1xT, bf1x, wf2xT, bf2x, gx1, bx1, out_x);
    k_ffn_ln_m64<<<EE / 64, 256, 0, stream>>>(e1, wf1eT, bf1e, wf2eT, bf2e, ge1, be1, out_e);
}
